// ClaimHeteroGNN_22935125361167
// MI455X (gfx1250) — matrix operand order repaired and checked by reading, not compiled, not run on hardware
//
#include <hip/hip_runtime.h>

// ---------------------------------------------------------------------------
// Types for CDNA5 WMMA
// ---------------------------------------------------------------------------
typedef __bf16    bf16x16 __attribute__((ext_vector_type(16)));
typedef float     f32x8   __attribute__((ext_vector_type(8)));
typedef unsigned  u32x4   __attribute__((ext_vector_type(4)));

// fp32 -> bf16 round-to-nearest-even
static __device__ __forceinline__ unsigned short f2bf(float f) {
    unsigned u = __float_as_uint(f);
    unsigned r = u + 0x7FFFu + ((u >> 16) & 1u);
    return (unsigned short)(r >> 16);
}
static __device__ __forceinline__ unsigned pack2bf(float lo, float hi) {
    return (unsigned)f2bf(lo) | ((unsigned)f2bf(hi) << 16);
}

// monotonic float<->uint encoding for atomicMax-based segment max
static __device__ __forceinline__ unsigned encf(float f) {
    unsigned u = __float_as_uint(f);
    return (u >> 31) ? ~u : (u | 0x80000000u);
}
static __device__ __forceinline__ float decf(unsigned k) {
    return (k >> 31) ? __uint_as_float(k & 0x7FFFFFFFu) : __uint_as_float(~k);
}

// ---------------------------------------------------------------------------
// Direct-fragment WMMA GEMM, no LDS:
//   C[M,N](fp32) = op( A[M,K](bf16) @ Bt[N,K](bf16, pre-transposed) + bias [+C] )
// Block: 128 threads (4 waves as 2x2), block tile 64M x 128N,
// wave tile 32M x 64N = 2x4 WMMA 16x16 tiles, BK=32.
// A fragment (16x32 bf16): lane half-groups cover K kb..kb+15, kb=16*(lane>=16)
//   -> two contiguous 16B chunks per lane.
// B fragment (32x16 bf16, stored as Bt rows): lane half-groups cover K kb..kb+15, kb=16*(lane>=16)
//   -> one contiguous 32B chunk per lane (two b128 loads).
// Requires K % 32 == 0, N % 128 == 0.
// ---------------------------------------------------------------------------
__global__ __launch_bounds__(128)
void gemm_wmma_direct(const unsigned short* __restrict__ Abf,  // [M,K] bf16
                      const unsigned short* __restrict__ Bt,   // [N,K] bf16
                      const float* __restrict__ bias,          // [N] or nullptr
                      float* __restrict__ C,                   // [M,N] fp32
                      unsigned short* __restrict__ Cbf,        // [M,N] bf16 dual out or nullptr
                      int M, int K, int N, int accumFlag, int reluFlag)
{
    const int lane  = threadIdx.x & 31;
    const int wave  = threadIdx.x >> 5;
    const int mTile = blockIdx.y * 64  + (wave >> 1) * 32;
    const int nTile = blockIdx.x * 128 + (wave & 1) * 64;
    const int fm    = lane & 15;
    const int kbA   = (lane >> 4) << 4;    // 0 or 16
    const int kbB   = (lane >> 4) << 4;    // 0 or 16

    int mr0 = mTile + fm;      if (mr0 >= M) mr0 = M - 1;
    int mr1 = mTile + 16 + fm; if (mr1 >= M) mr1 = M - 1;
    const unsigned short* a0 = Abf + (size_t)mr0 * K + kbA;
    const unsigned short* a1 = Abf + (size_t)mr1 * K + kbA;
    const unsigned short* b0 = Bt + (size_t)(nTile +  0 + fm) * K + kbB;
    const unsigned short* b1 = Bt + (size_t)(nTile + 16 + fm) * K + kbB;
    const unsigned short* b2 = Bt + (size_t)(nTile + 32 + fm) * K + kbB;
    const unsigned short* b3 = Bt + (size_t)(nTile + 48 + fm) * K + kbB;

    f32x8 acc[2][4] = {};
    union Frag { u32x4 q[2]; bf16x16 v; };

    for (int kt = 0; kt < K; kt += 32) {
        Frag A0, A1, B0, B1, B2, B3;
        A0.q[0] = *(const u32x4*)(a0 + kt);
        A0.q[1] = *(const u32x4*)(a0 + kt + 8);
        A1.q[0] = *(const u32x4*)(a1 + kt);
        A1.q[1] = *(const u32x4*)(a1 + kt + 8);
        B0.q[0] = *(const u32x4*)(b0 + kt);
        B0.q[1] = *(const u32x4*)(b0 + kt + 8);
        B1.q[0] = *(const u32x4*)(b1 + kt);
        B1.q[1] = *(const u32x4*)(b1 + kt + 8);
        B2.q[0] = *(const u32x4*)(b2 + kt);
        B2.q[1] = *(const u32x4*)(b2 + kt + 8);
        B3.q[0] = *(const u32x4*)(b3 + kt);
        B3.q[1] = *(const u32x4*)(b3 + kt + 8);

        acc[0][0] = __builtin_amdgcn_wmma_f32_16x16x32_bf16(false, A0.v, false, B0.v, (short)0, acc[0][0], false, false);
        acc[0][1] = __builtin_amdgcn_wmma_f32_16x16x32_bf16(false, A0.v, false, B1.v, (short)0, acc[0][1], false, false);
        acc[0][2] = __builtin_amdgcn_wmma_f32_16x16x32_bf16(false, A0.v, false, B2.v, (short)0, acc[0][2], false, false);
        acc[0][3] = __builtin_amdgcn_wmma_f32_16x16x32_bf16(false, A0.v, false, B3.v, (short)0, acc[0][3], false, false);
        acc[1][0] = __builtin_amdgcn_wmma_f32_16x16x32_bf16(false, A1.v, false, B0.v, (short)0, acc[1][0], false, false);
        acc[1][1] = __builtin_amdgcn_wmma_f32_16x16x32_bf16(false, A1.v, false, B1.v, (short)0, acc[1][1], false, false);
        acc[1][2] = __builtin_amdgcn_wmma_f32_16x16x32_bf16(false, A1.v, false, B2.v, (short)0, acc[1][2], false, false);
        acc[1][3] = __builtin_amdgcn_wmma_f32_16x16x32_bf16(false, A1.v, false, B3.v, (short)0, acc[1][3], false, false);
    }

    // D layout per 16x16 tile: row = r + 8*(lane>=16), col = lane&15
    #pragma unroll
    for (int sm = 0; sm < 2; ++sm) {
        int rowBase = mTile + sm * 16 + ((lane >> 4) << 3);
        #pragma unroll
        for (int sn = 0; sn < 4; ++sn) {
            int col = nTile + sn * 16 + (lane & 15);
            float bv = bias ? bias[col] : 0.0f;
            #pragma unroll
            for (int r = 0; r < 8; ++r) {
                int row = rowBase + r;
                if (row < M) {
                    size_t o = (size_t)row * N + col;
                    float v = acc[sm][sn][r] + bv;
                    if (accumFlag) v += C[o];
                    if (reluFlag)  v = fmaxf(v, 0.0f);
                    C[o] = v;
                    if (Cbf) Cbf[o] = f2bf(v);
                }
            }
        }
    }
}

// ---------------------------------------------------------------------------
// Conversion kernels
// ---------------------------------------------------------------------------
// fp32 [rows,K] -> bf16 same layout, optional per-row scale (mean divide). K even.
__global__ void cvt_f32_bf16(const float* __restrict__ in, const float* __restrict__ rowScale,
                             unsigned short* __restrict__ out, long long nPairs, int K)
{
    long long p = (long long)blockIdx.x * blockDim.x + threadIdx.x;
    if (p >= nPairs) return;
    long long i = p * 2;
    float s = rowScale ? rowScale[i / K] : 1.0f;
    float x = in[i] * s, y = in[i + 1] * s;
    *(unsigned*)(out + i) = pack2bf(x, y);
}

// W[K,N] fp32 -> Wt[N,K] bf16 (write-coalesced along K)
__global__ void wtrans_bf16(const float* __restrict__ W, unsigned short* __restrict__ Wt,
                            int K, int N)
{
    int idx = blockIdx.x * blockDim.x + threadIdx.x;
    if (idx >= K * N) return;
    int n = idx / K, k = idx - n * K;
    Wt[idx] = f2bf(W[(size_t)k * N + n]);
}

// ---------------------------------------------------------------------------
// Glue kernels
// ---------------------------------------------------------------------------
__global__ void fill_f32(float* p, float v, long long n) {
    long long i = (long long)blockIdx.x * blockDim.x + threadIdx.x;
    if (i < n) p[i] = v;
}
__global__ void fill_u32(unsigned* p, unsigned v, long long n) {
    long long i = (long long)blockIdx.x * blockDim.x + threadIdx.x;
    if (i < n) p[i] = v;
}

// one wave32 per edge: acc[dst] += x[src]; cnt[dst] += 1
__global__ void edge_scatter(const float* __restrict__ xsrc, const int* __restrict__ src,
                             const int* __restrict__ dst, float* __restrict__ acc,
                             float* __restrict__ cnt, int E)
{
    int gid = blockIdx.x * blockDim.x + threadIdx.x;
    int e = gid >> 5, lane = gid & 31;
    if (e >= E) return;
    int s = src[e], d = dst[e];
    const float* xp = xsrc + (size_t)s * 256;
    float* ap = acc + (size_t)d * 256;
    #pragma unroll
    for (int h = lane; h < 256; h += 32) atomicAdd(ap + h, xp[h]);
    if (lane == 0) atomicAdd(cnt + d, 1.0f);
}

__global__ void inv_count(float* cnt, int n) {
    int i = blockIdx.x * blockDim.x + threadIdx.x;
    if (i < n) cnt[i] = 1.0f / fmaxf(cnt[i], 1.0f);
}

// wave32 per sentence: scores[i] = dot(U[i], C2[batch[i]]); segment max via atomicMax
__global__ void attn_scores(const float* __restrict__ U, const float* __restrict__ C2,
                            const int* __restrict__ batch, float* __restrict__ scores,
                            unsigned* __restrict__ smax, int Ns)
{
    int gid = blockIdx.x * blockDim.x + threadIdx.x;
    int i = gid >> 5, lane = gid & 31;
    if (i >= Ns) return;
    int b = batch[i];
    const float* u = U + (size_t)i * 256;
    const float* c = C2 + (size_t)b * 256;
    float v = 0.f;
    #pragma unroll
    for (int h = lane; h < 256; h += 32) v += u[h] * c[h];
    #pragma unroll
    for (int o = 16; o > 0; o >>= 1) v += __shfl_xor(v, o, 32);
    if (lane == 0) { scores[i] = v; atomicMax(smax + b, encf(v)); }
}

__global__ void attn_exp(const float* __restrict__ scores, const int* __restrict__ batch,
                         const unsigned* __restrict__ smax, float* __restrict__ ebuf,
                         float* __restrict__ denom, int Ns)
{
    int i = blockIdx.x * blockDim.x + threadIdx.x;
    if (i >= Ns) return;
    int b = batch[i];
    float e = expf(scores[i] - decf(smax[b]));
    ebuf[i] = e;
    atomicAdd(denom + b, e);
}

// wave32 per sentence: g[batch[i]] += (e/denom) * xs[i]
__global__ void attn_pool(const float* __restrict__ xs, const float* __restrict__ ebuf,
                          const float* __restrict__ denom, const int* __restrict__ batch,
                          float* __restrict__ g, int Ns)
{
    int gid = blockIdx.x * blockDim.x + threadIdx.x;
    int i = gid >> 5, lane = gid & 31;
    if (i >= Ns) return;
    int b = batch[i];
    float a = ebuf[i] / denom[b];
    const float* xp = xs + (size_t)i * 256;
    float* gp = g + (size_t)b * 256;
    #pragma unroll
    for (int h = lane; h < 256; h += 32) atomicAdd(gp + h, a * xp[h]);
}

__global__ void build_z(const float* __restrict__ ch, const float* __restrict__ g,
                        float* __restrict__ z, int B)
{
    int idx = blockIdx.x * blockDim.x + threadIdx.x;
    if (idx >= B * 256) return;
    int b = idx >> 8, h = idx & 255;
    float c = ch[idx], gg = g[idx];
    float* zr = z + (size_t)b * 1024;
    zr[h] = c; zr[256 + h] = gg; zr[512 + h] = fabsf(c - gg); zr[768 + h] = c * gg;
}

__global__ void final_fc(const float* __restrict__ h, const float* __restrict__ W2,
                         const float* __restrict__ b2, float* __restrict__ out, int B)
{
    int idx = blockIdx.x * blockDim.x + threadIdx.x;
    if (idx >= B * 2) return;
    int b = idx >> 1, j = idx & 1;
    const float* hr = h + (size_t)b * 256;
    float acc = b2[j];
    for (int k = 0; k < 256; ++k) acc += hr[k] * W2[k * 2 + j];
    out[idx] = acc;
}

// ---------------------------------------------------------------------------
// Host orchestration
// ---------------------------------------------------------------------------
typedef unsigned short ushort_t;

static inline void launch_gemm(const ushort_t* A, const ushort_t* Bt, const float* bias,
                               float* C, ushort_t* Cbf,
                               int M, int K, int N, int accum, int relu, hipStream_t s)
{
    dim3 grid(N / 128, (M + 63) / 64);
    gemm_wmma_direct<<<grid, dim3(128), 0, s>>>(A, Bt, bias, C, Cbf, M, K, N, accum, relu);
}

extern "C" void kernel_launch(void* const* d_in, const int* in_sizes, int n_in,
                              void* d_out, int out_size, void* d_ws, size_t ws_size,
                              hipStream_t stream)
{
    const float* x_s     = (const float*)d_in[0];
    const float* x_n     = (const float*)d_in[1];
    const float* claim   = (const float*)d_in[2];
    const int*   batch_s = (const int*)  d_in[3];
    const int*   src_ss  = (const int*)  d_in[4];
    const int*   dst_ss  = (const int*)  d_in[5];
    const int*   src_sn  = (const int*)  d_in[6];
    const int*   dst_sn  = (const int*)  d_in[7];
    const int*   src_ns  = (const int*)  d_in[8];
    const int*   dst_ns  = (const int*)  d_in[9];
    const float* W_ps = (const float*)d_in[10]; const float* b_ps = (const float*)d_in[11];
    const float* W_pn = (const float*)d_in[12]; const float* b_pn = (const float*)d_in[13];
    const float* W_pc = (const float*)d_in[14]; const float* b_pc = (const float*)d_in[15];
    const float* conv_Wl = (const float*)d_in[16];
    const float* conv_bl = (const float*)d_in[17];
    const float* conv_Wr = (const float*)d_in[18];
    const float* W_attc = (const float*)d_in[19];
    const float* W_atts = (const float*)d_in[20];
    const float* W1 = (const float*)d_in[21]; const float* b1 = (const float*)d_in[22];
    const float* W2 = (const float*)d_in[23]; const float* b2 = (const float*)d_in[24];

    const int H = 256, D = 768;
    const int Ns  = in_sizes[3];
    const int Ne  = in_sizes[1] / D;
    const int B   = in_sizes[2] / D;
    const int Ess = in_sizes[4], Esn = in_sizes[6], Ens = in_sizes[8];
    const int L   = in_sizes[16] / (3 * H * H);

    // ---- workspace carve-out (byte based, 256B aligned) ----
    char* basep = (char*)d_ws;
    size_t off = 0;
    auto takeb = [&](size_t bytes) -> void* {
        void* p = basep + off;
        off = (off + bytes + 255) & ~(size_t)255;
        return p;
    };
    auto takef = [&](size_t n) -> float*    { return (float*)   takeb(n * 4); };
    auto takeh = [&](size_t n) -> ushort_t* { return (ushort_t*)takeb(n * 2); };

    // fp32 state
    float* xsA  = takef((size_t)Ns * H);
    float* xsB  = takef((size_t)Ns * H);
    float* xnA  = takef((size_t)Ne * H);
    float* xnB  = takef((size_t)Ne * H);
    float* aggS = takef((size_t)Ns * H);   // reused as U for attention
    float* aggN = takef((size_t)Ne * H);
    float* cntS = takef((size_t)Ns);
    float* cntN = takef((size_t)Ne);
    float* chb  = takef((size_t)B * H);
    float* c2   = takef((size_t)B * H);
    float* scor = takef((size_t)Ns);
    float* ebuf = takef((size_t)Ns);
    unsigned* smax = (unsigned*)takef((size_t)B);
    float* denm = takef((size_t)B);
    float* gbuf = takef((size_t)B * H);
    float* zbuf = takef((size_t)B * 4 * H);
    float* hbuf = takef((size_t)B * H);

    // bf16 activations
    ushort_t* xinS   = takeh((size_t)Ns * D);
    ushort_t* xinN   = takeh((size_t)Ne * D);
    ushort_t* chin   = takeh((size_t)B * D);
    ushort_t* xsbfA  = takeh((size_t)Ns * H);
    ushort_t* xsbfB  = takeh((size_t)Ns * H);
    ushort_t* xnbfA  = takeh((size_t)Ne * H);
    ushort_t* xnbfB  = takeh((size_t)Ne * H);
    ushort_t* aggSbf = takeh((size_t)Ns * H);
    ushort_t* aggNbf = takeh((size_t)Ne * H);
    ushort_t* chbf   = takeh((size_t)B * H);
    ushort_t* zbf    = takeh((size_t)B * 4 * H);

    // bf16 pre-transposed weights [N][K]
    ushort_t* WpsT = takeh((size_t)H * D);
    ushort_t* WpnT = takeh((size_t)H * D);
    ushort_t* WpcT = takeh((size_t)H * D);
    ushort_t* WlT[2][3]; ushort_t* WrT[2][3];
    for (int l = 0; l < L && l < 2; ++l)
        for (int r = 0; r < 3; ++r) {
            WlT[l][r] = takeh((size_t)H * H);
            WrT[l][r] = takeh((size_t)H * H);
        }
    ushort_t* WattcT = takeh((size_t)H * H);
    ushort_t* WattsT = takeh((size_t)H * H);
    ushort_t* W1T    = takeh((size_t)H * 4 * H);
    (void)ws_size; (void)n_in; (void)out_size;

    auto zero = [&](float* p, size_t n) {
        fill_f32<<<(unsigned)((n + 255) / 256), 256, 0, stream>>>(p, 0.0f, (long long)n);
    };
    auto wtrans = [&](const float* W, ushort_t* Wt, int K, int N) {
        wtrans_bf16<<<(K * N + 255) / 256, 256, 0, stream>>>(W, Wt, K, N);
    };
    auto cvt = [&](const float* in, const float* rs, ushort_t* out, size_t n, int K) {
        long long pairs = (long long)(n / 2);
        cvt_f32_bf16<<<(unsigned)((pairs + 255) / 256), 256, 0, stream>>>(in, rs, out, pairs, K);
    };

    // ---- one-time (per launch) weight transpose+convert, ~3.5 MB total ----
    wtrans(W_ps, WpsT, D, H);
    wtrans(W_pn, WpnT, D, H);
    wtrans(W_pc, WpcT, D, H);
    for (int l = 0; l < L && l < 2; ++l)
        for (int r = 0; r < 3; ++r) {
            wtrans(conv_Wl + (size_t)(l * 3 + r) * H * H, WlT[l][r], H, H);
            wtrans(conv_Wr + (size_t)(l * 3 + r) * H * H, WrT[l][r], H, H);
        }
    wtrans(W_attc, WattcT, H, H);
    wtrans(W_atts, WattsT, H, H);
    wtrans(W1, W1T, 4 * H, H);

    // ---- input conversion to bf16 ----
    cvt(x_s,   nullptr, xinS, (size_t)Ns * D, D);
    cvt(x_n,   nullptr, xinN, (size_t)Ne * D, D);
    cvt(claim, nullptr, chin, (size_t)B * D, D);

    // ---- input projections (WMMA, fused bias + relu, dual fp32/bf16 out) ----
    launch_gemm(xinS, WpsT, b_ps, xsA, xsbfA, Ns, D, H, 0, 1, stream);
    launch_gemm(xinN, WpnT, b_pn, xnA, xnbfA, Ne, D, H, 0, 1, stream);
    launch_gemm(chin, WpcT, b_pc, chb, chbf,  B,  D, H, 0, 1, stream);

    float*    xsCur = xsA;   float*    xsNxt = xsB;
    float*    xnCur = xnA;   float*    xnNxt = xnB;
    ushort_t* xsbfC = xsbfA; ushort_t* xsbfN = xsbfB;
    ushort_t* xnbfC = xnbfA; ushort_t* xnbfN = xnbfB;

    for (int l = 0; l < L && l < 2; ++l) {
        const float* bl0 = conv_bl + (size_t)(l * 3 + 0) * H;
        const float* bl1 = conv_bl + (size_t)(l * 3 + 1) * H;
        const float* bl2 = conv_bl + (size_t)(l * 3 + 2) * H;

        // ---- sentence: relu(mean_ss@Wl0 + bl0 + xs@Wr0 + mean_ns@Wl2 + bl2 + xs@Wr2)
        zero(aggS, (size_t)Ns * H); zero(cntS, Ns);
        edge_scatter<<<((size_t)Ess * 32 + 255) / 256, 256, 0, stream>>>(
            xsCur, src_ss, dst_ss, aggS, cntS, Ess);
        inv_count<<<(Ns + 255) / 256, 256, 0, stream>>>(cntS, Ns);
        cvt(aggS, cntS, aggSbf, (size_t)Ns * H, H);   // mean divide folded in
        launch_gemm(aggSbf, WlT[l][0], bl0, xsNxt, nullptr, Ns, H, H, 0, 0, stream);
        launch_gemm(xsbfC,  WrT[l][0], nullptr, xsNxt, nullptr, Ns, H, H, 1, 0, stream);

        zero(aggS, (size_t)Ns * H); zero(cntS, Ns);
        edge_scatter<<<((size_t)Ens * 32 + 255) / 256, 256, 0, stream>>>(
            xnCur, src_ns, dst_ns, aggS, cntS, Ens);
        inv_count<<<(Ns + 255) / 256, 256, 0, stream>>>(cntS, Ns);
        cvt(aggS, cntS, aggSbf, (size_t)Ns * H, H);
        launch_gemm(aggSbf, WlT[l][2], bl2, xsNxt, nullptr, Ns, H, H, 1, 0, stream);
        launch_gemm(xsbfC,  WrT[l][2], nullptr, xsNxt, xsbfN, Ns, H, H, 1, 1, stream); // relu + bf16 out

        // ---- entity: relu(mean_sn@Wl1 + bl1 + xn@Wr1)
        zero(aggN, (size_t)Ne * H); zero(cntN, Ne);
        edge_scatter<<<((size_t)Esn * 32 + 255) / 256, 256, 0, stream>>>(
            xsCur, src_sn, dst_sn, aggN, cntN, Esn);
        inv_count<<<(Ne + 255) / 256, 256, 0, stream>>>(cntN, Ne);
        cvt(aggN, cntN, aggNbf, (size_t)Ne * H, H);
        launch_gemm(aggNbf, WlT[l][1], bl1, xnNxt, nullptr, Ne, H, H, 0, 0, stream);
        launch_gemm(xnbfC,  WrT[l][1], nullptr, xnNxt, xnbfN, Ne, H, H, 1, 1, stream); // relu + bf16 out

        float* t;
        t = xsCur; xsCur = xsNxt; xsNxt = t;
        t = xnCur; xnCur = xnNxt; xnNxt = t;
        ushort_t* th;
        th = xsbfC; xsbfC = xsbfN; xsbfN = th;
        th = xnbfC; xnbfC = xnbfN; xnbfN = th;
    }

    // ---- claim-conditioned attentive pooling ----
    launch_gemm(xsbfC, WattsT, nullptr, aggS /*U*/, nullptr, Ns, H, H, 0, 0, stream);
    launch_gemm(chbf,  WattcT, nullptr, c2,         nullptr, B,  H, H, 0, 0, stream);
    fill_u32<<<(B + 255) / 256, 256, 0, stream>>>(smax, 0x007FFFFFu /*enc(-inf)*/, B);
    zero(denm, B); zero(gbuf, (size_t)B * H);
    attn_scores<<<((size_t)Ns * 32 + 255) / 256, 256, 0, stream>>>(aggS, c2, batch_s, scor, smax, Ns);
    attn_exp<<<(Ns + 255) / 256, 256, 0, stream>>>(scor, batch_s, smax, ebuf, denm, Ns);
    attn_pool<<<((size_t)Ns * 32 + 255) / 256, 256, 0, stream>>>(xsCur, ebuf, denm, batch_s, gbuf, Ns);

    // ---- head ----
    build_z<<<(B * 256 + 255) / 256, 256, 0, stream>>>(chb, gbuf, zbuf, B);
    cvt(zbuf, nullptr, zbf, (size_t)B * 4 * H, 4 * H);
    launch_gemm(zbf, W1T, b1, hbuf, nullptr, B, 4 * H, H, 0, 1, stream); // relu
    final_fc<<<(B * 2 + 255) / 256, 256, 0, stream>>>(hbuf, W2, b2, (float*)d_out, B);
}
